// DGCNNSemSeg_7206955123039
// MI455X (gfx1250) — matrix operand order repaired and checked by reading, not compiled, not run on hardware
//
#include <hip/hip_runtime.h>
#include <hip/hip_bf16.h>

// ---------------------------------------------------------------------------
// DGCNN semseg on MI455X (gfx1250): all dense layers via v_wmma_f32_16x16x32_f16
// Activations f16 point-major [B*N][C]; f32 accumulate; wave32 fragments laid
// out in contiguous k order (A and B: lane half h holds k=16h..16h+15 of each 32-k step;
// C/D: lane=col, vgpr v -> row v+8*(lane>>4)).
// ---------------------------------------------------------------------------

typedef __attribute__((ext_vector_type(16))) _Float16 v16h;
typedef __attribute__((ext_vector_type(8)))  _Float16 v8h;
typedef __attribute__((ext_vector_type(8)))  float    v8f;

#define WMMA(a, b, c) __builtin_amdgcn_wmma_f32_16x16x32_f16( \
    false, (a), false, (b), (short)0, (c), false, false)

static constexpr int NPTS  = 4096;
static constexpr int BATCH = 8;
static constexpr int TOTP  = BATCH * NPTS;   // 32768 points
static constexpr int KNN   = 20;
static constexpr int CATC  = 192;

// A-matrix fragment (16x32 f16): lane<16 holds M=lane, K=0..15;
// lane>=16 holds K=16..31.
__device__ __forceinline__ v16h load_a_frag(const _Float16* base, int pitch,
                                            int lane, int kbase) {
  int m = lane & 15, g = lane >> 4;
  v8h lo = *(const v8h*)(base + m * pitch + kbase + 16 * g);
  v8h hi = *(const v8h*)(base + m * pitch + kbase + 16 * g + 8);
  return __builtin_shufflevector(lo, hi, 0, 1, 2, 3, 4, 5, 6, 7,
                                 8, 9, 10, 11, 12, 13, 14, 15);
}

// B-matrix fragment (32x16 f16): lane holds column N=lane&15,
// K = 16*(lane>>4) .. +15 contiguous (storage is [col][K]).
__device__ __forceinline__ v16h load_b_frag(const _Float16* base, int pitch,
                                            int lane, int kbase) {
  int n = lane & 15, g = lane >> 4;
  return *(const v16h*)(base + (long)n * pitch + kbase + 16 * g);
}

// ---------------------------------------------------------------------------
// Pack points (B,9,N) f32 -> xin [B*N][16] f16 (zero padded)
// ---------------------------------------------------------------------------
__global__ __launch_bounds__(256)
void k_pack_points(const float* __restrict__ pts, _Float16* __restrict__ xin) {
  int p = blockIdx.x * 256 + threadIdx.x;
  if (p >= TOTP) return;
  int b = p >> 12, n = p & (NPTS - 1);
  _Float16* dst = xin + (long)p * 16;
  #pragma unroll
  for (int c = 0; c < 9; ++c)
    dst[c] = (_Float16)pts[((long)b * 9 + c) * NPTS + n];
  #pragma unroll
  for (int c = 9; c < 16; ++c) dst[c] = (_Float16)0.f;
}

// ---------------------------------------------------------------------------
// Fused EdgeConv: gather -> [ctr, nbr-ctr] -> 2x (WMMA GEMM + bias + relu)
// -> max over KNN -> cat slice.  4 points (80 cols) per workgroup, 4 waves.
// ---------------------------------------------------------------------------
template <int CIN, int KPAD>
__global__ __launch_bounds__(128)
void k_edgeconv(const _Float16* __restrict__ xin, int inPitch, int inOff,
                const int* __restrict__ knn,
                const float* __restrict__ w1, const float* __restrict__ b1,
                const float* __restrict__ w2, const float* __restrict__ b2,
                _Float16* __restrict__ cat, int outOff) {
  constexpr int CC   = 2 * CIN;
  constexpr int COLS = 4 * KNN;                    // 80
  constexpr int HP   = (KPAD > 64) ? KPAD : 64;    // sHin doubles as out2[col][64]
  __shared__ _Float16 sW1[64 * KPAD];
  __shared__ _Float16 sW2[64 * 64];
  __shared__ _Float16 sHin[COLS * HP];
  __shared__ _Float16 sH1[COLS * 64];

  const int tid = threadIdx.x;
  const int lane = tid & 31, wave = tid >> 5;
  const long pbase = (long)blockIdx.x * 4;

  for (int i = tid; i < 64 * KPAD; i += 128) {
    int r = i / KPAD, k = i - r * KPAD;
    sW1[i] = (k < CC) ? (_Float16)w1[r * CC + k] : (_Float16)0.f;
  }
  for (int i = tid; i < 64 * 64; i += 128) sW2[i] = (_Float16)w2[i];

  // Build edge features: rows 0..CIN-1 = ctr, rows CIN..2CIN-1 = nbr - ctr.
  for (int col = tid; col < COLS; col += 128) {
    int p = col / KNN, k = col - p * KNN;
    long P = pbase + p;
    int b = (int)(P >> 12), n = (int)(P & (NPTS - 1));
    int nidx = knn[((long)b * NPTS + n) * KNN + k];
    const _Float16* ctr = xin + ((long)b * NPTS + n) * inPitch + inOff;
    const _Float16* nbr = xin + ((long)b * NPTS + nidx) * inPitch + inOff;
    _Float16* dst = sHin + col * KPAD;
    #pragma unroll 8
    for (int c = 0; c < CIN; ++c) {
      _Float16 cv = ctr[c];
      dst[c] = cv;
      dst[CIN + c] = (_Float16)((float)nbr[c] - (float)cv);
    }
    for (int c = 2 * CIN; c < KPAD; ++c) dst[c] = (_Float16)0.f;
  }
  __syncthreads();

  // Layer 1: sH1[col][64] = relu(W1 @ Hin + b1)
  {
    constexpr int KS = KPAD / 32;
    const int m = wave;  // row tile 0..3
    v16h afr[KS];
    #pragma unroll
    for (int ks = 0; ks < KS; ++ks)
      afr[ks] = load_a_frag(sW1 + m * 16 * KPAD, KPAD, lane, ks * 32);
    #pragma unroll
    for (int ct = 0; ct < COLS / 16; ++ct) {
      v8f acc = {};
      #pragma unroll
      for (int ks = 0; ks < KS; ++ks) {
        v16h bf = load_b_frag(sHin + ct * 16 * KPAD, KPAD, lane, ks * 32);
        acc = WMMA(afr[ks], bf, acc);
      }
      int col = ct * 16 + (lane & 15), g = lane >> 4;
      v8h pk;
      #pragma unroll
      for (int v = 0; v < 8; ++v) {
        float x = acc[v] + b1[m * 16 + 8 * g + v];
        pk[v] = (_Float16)(x > 0.f ? x : 0.f);
      }
      *(v8h*)(sH1 + col * 64 + m * 16 + 8 * g) = pk;
    }
  }
  __syncthreads();

  // Layer 2: out2[col][64] = relu(W2 @ H1 + b2), reusing sHin storage.
  {
    const int m = wave;
    v16h afr[2];
    #pragma unroll
    for (int ks = 0; ks < 2; ++ks)
      afr[ks] = load_a_frag(sW2 + m * 16 * 64, 64, lane, ks * 32);
    #pragma unroll
    for (int ct = 0; ct < COLS / 16; ++ct) {
      v8f acc = {};
      #pragma unroll
      for (int ks = 0; ks < 2; ++ks) {
        v16h bf = load_b_frag(sH1 + ct * 16 * 64, 64, lane, ks * 32);
        acc = WMMA(afr[ks], bf, acc);
      }
      int col = ct * 16 + (lane & 15), g = lane >> 4;
      v8h pk;
      #pragma unroll
      for (int v = 0; v < 8; ++v) {
        float x = acc[v] + b2[m * 16 + 8 * g + v];
        pk[v] = (_Float16)(x > 0.f ? x : 0.f);
      }
      *(v8h*)(sHin + col * 64 + m * 16 + 8 * g) = pk;
    }
  }
  __syncthreads();

  // Max over KNN neighbors -> cat[point][outOff + f]
  for (int t = tid; t < 4 * 64; t += 128) {
    int p = t >> 6, f = t & 63;
    const _Float16* src = sHin + (p * KNN) * 64 + f;
    float mx = -1e30f;
    #pragma unroll
    for (int k = 0; k < KNN; ++k) {
      float v = (float)src[k * 64];
      mx = v > mx ? v : mx;
    }
    cat[(pbase + p) * CATC + outOff + f] = (_Float16)mx;
  }
}

// ---------------------------------------------------------------------------
// local = relu(W(1024x192) @ cat + b), fused column-max into gfeat[b][1024]
// via shfl_xor tile reduce + atomicMax on float bits (values >= 0).
// ---------------------------------------------------------------------------
__global__ __launch_bounds__(256)
void k_local_max(const float* __restrict__ W, const float* __restrict__ bias,
                 const _Float16* __restrict__ Bmat, float* __restrict__ gfeat) {
  constexpr int KD = CATC;
  __shared__ _Float16 sA[16 * KD];
  const int tid = threadIdx.x, lane = tid & 31, wave = tid >> 5;
  const int rowbase = blockIdx.y * 16;
  for (int i = tid; i < 16 * KD; i += 256) {
    int r = i / KD, k = i - r * KD;
    sA[i] = (_Float16)W[(long)(rowbase + r) * KD + k];
  }
  __syncthreads();

  const long colbase = (long)blockIdx.x * 256 + wave * 32;
  const _Float16* b0 = Bmat + colbase * CATC;
  v8f acc0 = {}, acc1 = {};
  #pragma unroll
  for (int ks = 0; ks < KD / 32; ++ks) {
    __builtin_prefetch(b0 + 32 * CATC + ks * 32, 0, 0);
    v16h a = load_a_frag(sA, KD, lane, ks * 32);
    v16h bf0 = load_b_frag(b0, CATC, lane, ks * 32);
    v16h bf1 = load_b_frag(b0 + 16 * CATC, CATC, lane, ks * 32);
    acc0 = WMMA(a, bf0, acc0);
    acc1 = WMMA(a, bf1, acc1);
  }
  const int g = lane >> 4;
  #pragma unroll
  for (int t = 0; t < 2; ++t) {
    v8f acc = t ? acc1 : acc0;
    int b = (int)((colbase + t * 16) >> 12);
    #pragma unroll
    for (int v = 0; v < 8; ++v) {
      int f = rowbase + 8 * g + v;
      float x = acc[v] + bias[f];
      x = x > 0.f ? x : 0.f;
      // max across the 16 columns of the tile (lanes 0..15 / 16..31 groups)
      x = fmaxf(x, __shfl_xor(x, 1, 32));
      x = fmaxf(x, __shfl_xor(x, 2, 32));
      x = fmaxf(x, __shfl_xor(x, 4, 32));
      x = fmaxf(x, __shfl_xor(x, 8, 32));
      if ((lane & 15) == 0)
        atomicMax((int*)&gfeat[(long)b * 1024 + f], __float_as_int(x));
    }
  }
}

__global__ __launch_bounds__(256)
void k_zero_f32(float* __restrict__ p, int n) {
  int i = blockIdx.x * 256 + threadIdx.x;
  if (i < n) p[i] = 0.f;
}

// gvec[b][o] = sum_c W_seg1[o][c<1024] * gfeat[b][c]   (tiny per-batch GEMV)
__global__ __launch_bounds__(256)
void k_gvec(const float* __restrict__ W, const float* __restrict__ gfeat,
            float* __restrict__ gvec) {
  __shared__ float sg[1024];
  int b = blockIdx.x;
  for (int i = threadIdx.x; i < 1024; i += 256) sg[i] = gfeat[(long)b * 1024 + i];
  __syncthreads();
  for (int o = threadIdx.x; o < 512; o += 256) {
    const float* wr = W + (long)o * 1216;
    float s = 0.f;
    for (int c = 0; c < 1024; ++c) s += wr[c] * sg[c];
    gvec[(long)b * 512 + o] = s;
  }
}

// ---------------------------------------------------------------------------
// Generic WMMA GEMM, f16 output: Out[p][rowbase..] = relu(W @ B + bias (+gadd))
// ---------------------------------------------------------------------------
template <int KDIM>
__global__ __launch_bounds__(256)
void k_gemm_h16(const float* __restrict__ W, int wPitch, int wOff,
                const float* __restrict__ bias,
                const float* __restrict__ gadd, int gPitch,
                const _Float16* __restrict__ Bmat, int bPitch,
                _Float16* __restrict__ Out, int outPitch) {
  __shared__ _Float16 sA[16 * KDIM];
  const int tid = threadIdx.x, lane = tid & 31, wave = tid >> 5;
  const int rowbase = blockIdx.y * 16;
  for (int i = tid; i < 16 * KDIM; i += 256) {
    int r = i / KDIM, k = i - r * KDIM;
    sA[i] = (_Float16)W[(long)(rowbase + r) * wPitch + wOff + k];
  }
  __syncthreads();

  const long colbase = (long)blockIdx.x * 256 + wave * 32;
  const _Float16* b0 = Bmat + colbase * bPitch;
  v8f acc0 = {}, acc1 = {};
  #pragma unroll
  for (int ks = 0; ks < KDIM / 32; ++ks) {
    __builtin_prefetch(b0 + 32 * bPitch + ks * 32, 0, 0);
    v16h a = load_a_frag(sA, KDIM, lane, ks * 32);
    v16h bf0 = load_b_frag(b0, bPitch, lane, ks * 32);
    v16h bf1 = load_b_frag(b0 + 16 * bPitch, bPitch, lane, ks * 32);
    acc0 = WMMA(a, bf0, acc0);
    acc1 = WMMA(a, bf1, acc1);
  }
  const int g = lane >> 4;
  #pragma unroll
  for (int t = 0; t < 2; ++t) {
    v8f acc = t ? acc1 : acc0;
    long p = colbase + t * 16 + (lane & 15);
    int b = (int)(p >> 12);
    v8h pk;
    #pragma unroll
    for (int v = 0; v < 8; ++v) {
      int f = rowbase + 8 * g + v;
      float x = acc[v] + bias[f];
      if (gadd) x += gadd[(long)b * gPitch + f];
      x = x > 0.f ? x : 0.f;
      pk[v] = (_Float16)x;
    }
    *(v8h*)(Out + p * outPitch + rowbase + 8 * g) = pk;
  }
}

// Final logits: (13x256) @ h + b -> f32 out[b][class][n]; rows 13..15 dropped.
__global__ __launch_bounds__(256)
void k_logit(const float* __restrict__ W, const float* __restrict__ bias,
             const _Float16* __restrict__ Bmat, float* __restrict__ out) {
  constexpr int KD = 256;
  __shared__ _Float16 sA[16 * KD];
  const int tid = threadIdx.x, lane = tid & 31, wave = tid >> 5;
  for (int i = tid; i < 16 * KD; i += 256) {
    int r = i / KD, k = i - r * KD;
    sA[i] = (r < 13) ? (_Float16)W[(long)r * KD + k] : (_Float16)0.f;
  }
  __syncthreads();

  const long colbase = (long)blockIdx.x * 256 + wave * 32;
  const _Float16* b0 = Bmat + colbase * KD;
  v8f acc0 = {}, acc1 = {};
  #pragma unroll
  for (int ks = 0; ks < KD / 32; ++ks) {
    v16h a = load_a_frag(sA, KD, lane, ks * 32);
    v16h bf0 = load_b_frag(b0, KD, lane, ks * 32);
    v16h bf1 = load_b_frag(b0 + 16 * KD, KD, lane, ks * 32);
    acc0 = WMMA(a, bf0, acc0);
    acc1 = WMMA(a, bf1, acc1);
  }
  const int g = lane >> 4;
  #pragma unroll
  for (int t = 0; t < 2; ++t) {
    v8f acc = t ? acc1 : acc0;
    long p = colbase + t * 16 + (lane & 15);
    int b = (int)(p >> 12), n = (int)(p & (NPTS - 1));
    #pragma unroll
    for (int v = 0; v < 8; ++v) {
      int f = 8 * g + v;
      if (f < 13)
        out[((long)b * 13 + f) * NPTS + n] = acc[v] + bias[f];
    }
  }
}

// ---------------------------------------------------------------------------
extern "C" void kernel_launch(void* const* d_in, const int* in_sizes, int n_in,
                              void* d_out, int out_size, void* d_ws,
                              size_t ws_size, hipStream_t stream) {
  const float* points  = (const float*)d_in[0];
  const int*   knn     = (const int*)d_in[1];
  const float* ec0w1 = (const float*)d_in[2],  *ec0b1 = (const float*)d_in[3];
  const float* ec0w2 = (const float*)d_in[4],  *ec0b2 = (const float*)d_in[5];
  const float* ec1w1 = (const float*)d_in[6],  *ec1b1 = (const float*)d_in[7];
  const float* ec1w2 = (const float*)d_in[8],  *ec1b2 = (const float*)d_in[9];
  const float* ec2w1 = (const float*)d_in[10], *ec2b1 = (const float*)d_in[11];
  const float* ec2w2 = (const float*)d_in[12], *ec2b2 = (const float*)d_in[13];
  const float* wloc  = (const float*)d_in[14], *bloc  = (const float*)d_in[15];
  const float* wseg1 = (const float*)d_in[16], *bseg1 = (const float*)d_in[17];
  const float* wseg2 = (const float*)d_in[18], *bseg2 = (const float*)d_in[19];
  const float* wlog  = (const float*)d_in[20], *blog  = (const float*)d_in[21];
  float* out = (float*)d_out;

  char* ws = (char*)d_ws;
  _Float16* xin   = (_Float16*)(ws);                    // 32768*16*2   = 1 MB
  _Float16* cat   = (_Float16*)(ws + (1l << 20));       // 32768*192*2  = 12 MB
  _Float16* hseg1 = (_Float16*)(ws + 13631488l);        // 32768*512*2  = 32 MB
  _Float16* hseg2 = (_Float16*)(ws + 47185920l);        // 32768*256*2  = 16 MB
  float*    gfeat = (float*)   (ws + 63963136l);        // 8*1024*4
  float*    gvec  = (float*)   (ws + 63995904l);        // 8*512*4

  // 1) pack points to f16 [point][16]
  k_pack_points<<<TOTP / 256, 256, 0, stream>>>(points, xin);

  // 2) three fused EdgeConv blocks -> cat[point][192]
  k_edgeconv<9, 32><<<TOTP / 4, 128, 0, stream>>>(
      xin, 16, 0, knn, ec0w1, ec0b1, ec0w2, ec0b2, cat, 0);
  k_edgeconv<64, 128><<<TOTP / 4, 128, 0, stream>>>(
      cat, CATC, 0, knn, ec1w1, ec1b1, ec1w2, ec1b2, cat, 64);
  k_edgeconv<64, 128><<<TOTP / 4, 128, 0, stream>>>(
      cat, CATC, 64, knn, ec2w1, ec2b1, ec2w2, ec2b2, cat, 128);

  // 3) local layer fused with global max -> gfeat[b][1024]
  k_zero_f32<<<(BATCH * 1024 + 255) / 256, 256, 0, stream>>>(gfeat, BATCH * 1024);
  k_local_max<<<dim3(TOTP / 256, 64), 256, 0, stream>>>(wloc, bloc, cat, gfeat);

  // 4) per-batch gvec = W_seg1[:, :1024] @ gfeat
  k_gvec<<<BATCH, 256, 0, stream>>>(wseg1, gfeat, gvec);

  // 5) seg1 = relu(W_seg1[:,1024:] @ cat + gvec + b1)  -> [point][512]
  k_gemm_h16<CATC><<<dim3(TOTP / 256, 32), 256, 0, stream>>>(
      wseg1, 1216, 1024, bseg1, gvec, 512, cat, CATC, hseg1, 512);

  // 6) seg2 = relu(W_seg2 @ seg1 + b2)  -> [point][256]
  k_gemm_h16<512><<<dim3(TOTP / 256, 16), 256, 0, stream>>>(
      wseg2, 512, 0, bseg2, nullptr, 0, hseg1, 512, hseg2, 256);

  // 7) logits -> f32 (B, 13, N)
  k_logit<<<dim3(TOTP / 256, 1), 256, 0, stream>>>(wlog, blog, hseg2, out);

  (void)in_sizes; (void)n_in; (void)out_size; (void)ws_size;
}
